// GatedPrototypeDistillationLoss_64372969832683
// MI455X (gfx1250) — matrix operand order repaired and checked by reading, not compiled, not run on hardware
//
#include <hip/hip_runtime.h>
#include <hip/hip_bf16.h>
#include <math.h>

// ---------------------------------------------------------------------------
// GatedPrototypeDistillationLoss for MI455X (gfx1250, wave32, WMMA)
//   B=32768 rows, D=512, K=2048 prototypes, T=0.07
//   sims GEMM via v_wmma_f32_16x16x32_bf16, fused softmax/argmax epilogue.
//   - B fragments pinned in registers via empty asm so loads clause together
//     and waits are incremental (no per-WMMA loadcnt-0 drain).
//   - Constant-shift softmax: z = sim/T in [-1/T, 1/T], so exp(z - 1/T) is
//     always in [e^{-2/T}, 1]; no online max rescaling needed.
// ---------------------------------------------------------------------------

#define NB   32768
#define ND   512
#define NK   2048
#define TEMP 0.07f

typedef __attribute__((ext_vector_type(16))) __bf16 v16bf;
typedef __attribute__((ext_vector_type(8)))  float  v8f;
typedef __attribute__((ext_vector_type(8)))  float  fv8;

// ---------------------------------------------------------------------------
// Kernel 1: L2-normalize prototypes [K, D] fp32 -> bf16 row-major
// ---------------------------------------------------------------------------
__global__ __launch_bounds__(256) void proto_norm_kernel(
    const float* __restrict__ gp, unsigned short* __restrict__ pnb_us)
{
    __bf16* pnb = (__bf16*)pnb_us;
    const int k = blockIdx.x;
    const int t = threadIdx.x;
    const float x0 = gp[(size_t)k * ND + t];
    const float x1 = gp[(size_t)k * ND + 256 + t];
    float ss = x0 * x0 + x1 * x1;
    #pragma unroll
    for (int off = 16; off >= 1; off >>= 1)
        ss += __shfl_xor(ss, off, 32);
    __shared__ float red[8];
    if ((t & 31) == 0) red[t >> 5] = ss;
    __syncthreads();
    float tot = 0.f;
    #pragma unroll
    for (int i = 0; i < 8; ++i) tot += red[i];
    const float scale = 1.0f / fmaxf(sqrtf(tot), 1e-12f);
    pnb[(size_t)k * ND + t]       = (__bf16)(x0 * scale);
    pnb[(size_t)k * ND + 256 + t] = (__bf16)(x1 * scale);
}

// ---------------------------------------------------------------------------
// Kernel 2: fused normalize(E) @ normalize(P)^T with softmax stats.
// 256 threads = 8 waves; each wave owns 16 rows; block owns 128 rows.
// grid = 32768/128 = 256 blocks.
// ---------------------------------------------------------------------------
__global__ __launch_bounds__(256, 1) void gemm_stats_kernel(
    const float* __restrict__ emb,
    const float* __restrict__ best_sim,
    const float* __restrict__ proto_conf,
    const unsigned short* __restrict__ pnb_us,
    float* __restrict__ gate_out,
    float* __restrict__ partials)
{
    const __bf16* pnb = (const __bf16*)pnb_us;
    const float invT = 1.0f / TEMP;
    const float ZC   = invT;            // constant softmax shift (max possible z)

    const int tid  = threadIdx.x;
    const int wave = tid >> 5;
    const int lane = tid & 31;
    const int half = lane >> 4;   // 0 or 1
    const int lrow = lane & 15;

    const int row_base = blockIdx.x * 128 + wave * 16;
    const int aRow     = row_base + lrow;
    const float* arow  = emb + (size_t)aRow * ND;
    const int koff     = half * 8;     // A half-wave K offset

    // ---- pass 1: row sum-of-squares (lanes l and l^16 cover the full row)
    float ss = 0.f;
    #pragma unroll
    for (int d = 0; d < 16; ++d) {
        const int k0 = d * 32 + koff;
        fv8 x = *(const fv8*)(arow + k0);
        fv8 y = *(const fv8*)(arow + k0 + 16);
        #pragma unroll
        for (int i = 0; i < 8; ++i) ss += x[i] * x[i] + y[i] * y[i];
    }
    ss += __shfl_xor(ss, 16, 32);
    const float scale = 1.0f / fmaxf(sqrtf(ss), 1e-12f);

    // ---- pass 2: build the full-K A fragment set in registers (128 VGPRs)
    v16bf afrag[16];
    #pragma unroll
    for (int d = 0; d < 16; ++d) {
        const int k0 = d * 32 + koff;
        fv8 x = *(const fv8*)(arow + k0);
        fv8 y = *(const fv8*)(arow + k0 + 16);
        #pragma unroll
        for (int i = 0; i < 8; ++i) {
            afrag[d][i]     = (__bf16)(x[i] * scale);
            afrag[d][i + 8] = (__bf16)(y[i] * scale);
        }
    }

    // ---- softmax / argmax state: slot j -> row (row_base + j + 8*half)
    float m[8], s1[8], s2[8];
    int   bidx[8];
    #pragma unroll
    for (int j = 0; j < 8; ++j) { m[j] = -INFINITY; s1[j] = 0.f; s2[j] = 0.f; bidx[j] = 0; }

    // ---- sweep all 2048 prototype columns in 16-wide tiles
    for (int nt = 0; nt < NK / 16; ++nt) {
        const int ncol = nt * 16 + lrow;                      // column this lane reduces
        const __bf16* bp = pnb + (size_t)ncol * ND + half * 8;

        // Load ALL 16 B fragments, then pin each one so the loads cannot be
        // rematerialized at the WMMA sites (forces a load clause + incremental
        // loadcnt waits, letting WMMA d overlap loads d+1..15).
        fv8 braw[16];
        #pragma unroll
        for (int d = 0; d < 16; ++d) {
            typedef __attribute__((ext_vector_type(4))) float fv4;
            const fv4 lo = *(const fv4*)(bp + d * 32);
            const fv4 hi = *(const fv4*)(bp + d * 32 + 16);
            braw[d] = __builtin_shufflevector(lo, hi, 0, 1, 2, 3, 4, 5, 6, 7);
        }
        #pragma unroll
        for (int d = 0; d < 16; ++d)
            asm volatile("" : "+v"(braw[d]));

        // Back-to-back accumulation chain on the matrix pipe.
        v8f acc = {};
        #pragma unroll
        for (int d = 0; d < 16; ++d)
            acc = __builtin_amdgcn_wmma_f32_16x16x32_bf16(
                false, afrag[d], false, __builtin_bit_cast(v16bf, braw[d]),
                (short)0, acc, false, false);

        // Constant-shift softmax accumulation + argmax (1 exp per element).
        #pragma unroll
        for (int j = 0; j < 8; ++j) {
            const float z = acc[j] * invT;
            const float e = __expf(z - ZC);
            s1[j] += e;
            s2[j] += z * e;
            if (z > m[j]) { m[j] = z; bidx[j] = ncol; }   // strict > = first-index argmax
        }
    }

    // ---- merge across the 16 lanes sharing each row (s1/s2 plain adds)
    float local = 0.f;
    const float inv_max_ent = 1.0f / (logf((float)NK) + 1e-8f);
    #pragma unroll
    for (int j = 0; j < 8; ++j) {
        float mj = m[j], s1j = s1[j], s2j = s2[j];
        int   ij = bidx[j];
        #pragma unroll
        for (int off = 1; off < 16; off <<= 1) {
            const float mo  = __shfl_xor(mj,  off, 32);
            const float s1o = __shfl_xor(s1j, off, 32);
            const float s2o = __shfl_xor(s2j, off, 32);
            const int   io  = __shfl_xor(ij,  off, 32);
            s1j += s1o;
            s2j += s2o;
            ij  = (mo > mj) ? io : ((mj > mo) ? ij : (io < ij ? io : ij));
            mj  = fmaxf(mj, mo);
        }
        if (lrow == 0) {
            const int r = row_base + j + 8 * half;
            const float logZ    = ZC + logf(s1j);
            const float entropy = logZ - s2j / s1j;           // nats
            const float tau     = 0.5f - 0.2f * (entropy * inv_max_ent);
            const float conf    = proto_conf[ij];
            const float sg      = 1.0f / (1.0f + __expf(-(best_sim[r] - tau) * invT));
            const float gate    = conf * sg;
            const float smax    = mj * TEMP;                  // back to sim space
            const float dist    = sqrtf(fmaxf(0.f, 2.0f - 2.0f * smax));
            gate_out[r] = gate;
            local += gate * dist;
        }
    }

    // ---- block reduction of gate*dist partial
    __shared__ float sred[256];
    sred[tid] = (lrow == 0) ? local : 0.f;
    __syncthreads();
    #pragma unroll
    for (int s = 128; s >= 1; s >>= 1) {
        if (tid < s) sred[tid] += sred[tid + s];
        __syncthreads();
    }
    if (tid == 0) partials[blockIdx.x] = sred[0];
}

// ---------------------------------------------------------------------------
// Kernel 3: final reduction of 256 block partials -> L_proto = sum / B
// ---------------------------------------------------------------------------
__global__ __launch_bounds__(256) void finalize_kernel(
    const float* __restrict__ partials, float* __restrict__ out)
{
    __shared__ float sred[256];
    const int t = threadIdx.x;
    sred[t] = partials[t];
    __syncthreads();
    #pragma unroll
    for (int s = 128; s >= 1; s >>= 1) {
        if (t < s) sred[t] += sred[t + s];
        __syncthreads();
    }
    if (t == 0) out[0] = sred[0] * (1.0f / (float)NB);
}

// ---------------------------------------------------------------------------
extern "C" void kernel_launch(void* const* d_in, const int* in_sizes, int n_in,
                              void* d_out, int out_size, void* d_ws, size_t ws_size,
                              hipStream_t stream)
{
    const float* emb   = (const float*)d_in[0];   // [B, D]
    const float* bsim  = (const float*)d_in[1];   // [B]
    const float* gp    = (const float*)d_in[2];   // [K, D]
    const float* conf  = (const float*)d_in[3];   // [K]

    float* out = (float*)d_out;                   // [0]=L_proto, [1..B]=gate

    unsigned short* pnb = (unsigned short*)d_ws;                       // K*D bf16 = 2 MB
    float* partials = (float*)((char*)d_ws + (size_t)NK * ND * sizeof(unsigned short));

    proto_norm_kernel<<<NK, 256, 0, stream>>>(gp, pnb);
    gemm_stats_kernel<<<NB / 128, 256, 0, stream>>>(emb, bsim, conf, pnb,
                                                    out + 1, partials);
    finalize_kernel<<<1, 256, 0, stream>>>(partials, out);
}
